// MultiTaskGAT_592705487421
// MI455X (gfx1250) — compile-verified, matrix operand order checked by reading, not run on hardware
//
#include <hip/hip_runtime.h>
#include <hip/hip_bf16.h>

typedef __attribute__((ext_vector_type(16))) __bf16 v16bf;
typedef __attribute__((ext_vector_type(8)))  float  v8f;

// ---------------------------------------------------------------------------
// Generic WMMA GEMM:  C[M, ncols] = A[rowidx[m], 0:K] @ W[K, ncols] (+ bias)
// block = 128 threads (4 waves); each wave computes one 16x16 output tile.
// grid.x = ceil(M/64) row tiles, grid.y = ceil(ncols/16) col tiles.
// A, W are fp32 in memory; converted to bf16 ONCE at LDS staging time.
// lA is [64][K] bf16 row-major; lW is the 16-col weight tile TRANSPOSED to
// [16][K] bf16 so both A and B fragments are two contiguous 16B LDS reads.
// ---------------------------------------------------------------------------
template <int K>
__global__ __launch_bounds__(128) void gemm_wmma(
    const float* __restrict__ A, const int* __restrict__ rowidx,
    const float* __restrict__ W, const float* __restrict__ bias,
    float* __restrict__ C, int M, int ncols)
{
    constexpr int LDA = K + 8;                    // bf16 elems; keeps 16B row align,
                                                  // pitch 4*row banks (conflict-free)
    __shared__ alignas(16) __bf16 lA[64 * LDA];
    __shared__ alignas(16) __bf16 lW[16 * LDA];

    const int tid     = threadIdx.x;
    const int rowTile = blockIdx.x * 64;
    const int j0      = blockIdx.y * 16;

    // Prefetch next row tile of A (streaming dimension) into L2.
    {
        int pr = rowTile + 64 + tid;              // one row head per thread
        if (pr < M && tid < 64) {
            int ar = rowidx ? rowidx[pr] : pr;
            __builtin_prefetch(A + (size_t)ar * K, 0, 1);
        }
    }

    // Cooperative load of 64 x K activation tile (optionally gathered), cvt->bf16.
    for (int idx = tid; idx < 64 * K; idx += 128) {
        int r = idx / K, c = idx % K;
        int gr = rowTile + r;
        float v = 0.f;
        if (gr < M) {
            int ar = rowidx ? rowidx[gr] : gr;
            v = A[(size_t)ar * K + c];
        }
        lA[r * LDA + c] = (__bf16)v;
    }
    // Cooperative load of K x 16 weight tile; store transposed [j][k] as bf16.
    for (int idx = tid; idx < K * 16; idx += 128) {
        int k = idx / 16, j = idx % 16;           // consecutive tid -> coalesced gmem
        int gc = j0 + j;
        float v = (gc < ncols) ? W[(size_t)k * ncols + gc] : 0.f;
        lW[j * LDA + k] = (__bf16)v;
    }
    __syncthreads();

    const int w    = tid >> 5;                    // wave id 0..3 -> rows w*16..+15
    const int l    = tid & 31;                    // lane
    const int mr   = l & 15;                      // M (A) / N (B) within tile
    const int half = l >> 4;                      // lane half selects K sub-block
    const int kb   = half * 8;

    union Frag { uint4 q[2]; v16bf v; };
    const __bf16* pa = lA + (w * 16 + mr) * LDA;  // A row m
    const __bf16* pb = lW + mr * LDA;             // B col n (transposed row)

    v8f acc = {};
#pragma unroll
    for (int s = 0; s < K; s += 32) {
        Frag fa, fb;
        fa.q[0] = *(const uint4*)(pa + s + kb);        // K = s+kb .. +7
        fa.q[1] = *(const uint4*)(pa + s + 16 + kb);   // K = s+16+kb .. +7
        fb.q[0] = *(const uint4*)(pb + s + kb);
        fb.q[1] = *(const uint4*)(pb + s + 16 + kb);
        acc = __builtin_amdgcn_wmma_f32_16x16x32_bf16(
                  false, fa.v, false, fb.v, (short)0, acc, false, false);
    }

    const int col = j0 + mr;                      // D: N = lane&15, M = vgpr+8*half
    if (col < ncols) {
        const float bv = bias ? bias[col] : 0.f;
        const int rbase = rowTile + w * 16 + half * 8;
#pragma unroll
        for (int r = 0; r < 8; ++r) {
            int row = rbase + r;
            if (row < M) C[(size_t)row * ncols + col] = acc[r] + bv;
        }
    }
}

// ---------------------------------------------------------------------------
// GATv2 edge message + logits + segment-max (sortable-uint atomicMax).
// one thread per (edge, head); HID=128, H=8, C=16 hardcoded.
// ---------------------------------------------------------------------------
__global__ void edge_attn_logits(
    const float* __restrict__ xl, const float* __restrict__ xr,
    const float* __restrict__ em, const int* __restrict__ src,
    const int* __restrict__ dst, const float* __restrict__ att,
    float* __restrict__ logits, unsigned* __restrict__ mxbits, int E)
{
    int idx = blockIdx.x * blockDim.x + threadIdx.x;
    if (idx >= E * 8) return;
    int e = idx >> 3, hh = idx & 7;
    int s = src[e], d = dst[e];
    const float* pl = xl + (size_t)s * 128 + hh * 16;
    const float* pr = xr + (size_t)d * 128 + hh * 16;
    const float* pe = em + (size_t)e * 128 + hh * 16;
    const float* pa = att + hh * 16;
    float acc = 0.f;
#pragma unroll
    for (int c = 0; c < 16; ++c) {
        float m = pl[c] + pr[c] + pe[c];
        m = (m > 0.f) ? m : 0.2f * m;             // LeakyReLU(0.2)
        acc += m * pa[c];
    }
    logits[idx] = acc;
    unsigned u = __float_as_uint(acc);
    unsigned key = (u & 0x80000000u) ? ~u : (u | 0x80000000u);
    atomicMax(mxbits + (size_t)d * 8 + hh, key);
}

__global__ void edge_exp_sum(
    float* __restrict__ logits, const int* __restrict__ dst,
    const unsigned* __restrict__ mxbits, float* __restrict__ den, int E)
{
    int idx = blockIdx.x * blockDim.x + threadIdx.x;
    if (idx >= E * 8) return;
    int e = idx >> 3, hh = idx & 7;
    int d = dst[e];
    unsigned key = mxbits[(size_t)d * 8 + hh];
    float mx = (key & 0x80000000u) ? __uint_as_float(key ^ 0x80000000u)
                                   : __uint_as_float(~key);
    float ev = expf(logits[idx] - mx);
    logits[idx] = ev;                             // overwrite logits with exp
    atomicAdd(den + (size_t)d * 8 + hh, ev);
}

__global__ void edge_aggregate(
    const float* __restrict__ ev, const float* __restrict__ den,
    const float* __restrict__ xl, const int* __restrict__ src,
    const int* __restrict__ dst, float* __restrict__ outacc, int E)
{
    int idx = blockIdx.x * blockDim.x + threadIdx.x;
    if (idx >= E * 128) return;                   // 102.4M < 2^31
    int e = idx >> 7, f = idx & 127;
    int hh = f >> 4;
    int s = src[e], d = dst[e];
    float alpha = ev[(size_t)e * 8 + hh] / den[(size_t)d * 8 + hh];
    atomicAdd(outacc + (size_t)d * 128 + f, alpha * xl[(size_t)s * 128 + f]);
}

// out = relu(LayerNorm(acc + bias) * ls + lb); one 128-thread block per node.
__global__ __launch_bounds__(128) void node_ln_relu(
    const float* __restrict__ acc, const float* __restrict__ bias,
    const float* __restrict__ ls, const float* __restrict__ lb,
    float* __restrict__ hout)
{
    __shared__ float red[128];
    const int node = blockIdx.x, t = threadIdx.x;
    float v = acc[(size_t)node * 128 + t] + bias[t];
    red[t] = v; __syncthreads();
    for (int s = 64; s > 0; s >>= 1) { if (t < s) red[t] += red[t + s]; __syncthreads(); }
    float mu = red[0] / 128.f; __syncthreads();
    float dv = v - mu;
    red[t] = dv * dv; __syncthreads();
    for (int s = 64; s > 0; s >>= 1) { if (t < s) red[t] += red[t + s]; __syncthreads(); }
    float var = red[0] / 128.f;
    float o = dv * rsqrtf(var + 1e-5f) * ls[t] + lb[t];
    hout[(size_t)node * 128 + t] = fmaxf(o, 0.f);
}

template <int NC>
__global__ void softmax_rows(const float* __restrict__ in,
                             float* __restrict__ outp, int M)
{
    int r = blockIdx.x * blockDim.x + threadIdx.x;
    if (r >= M) return;
    float v[NC], mx = -3.4e38f;
#pragma unroll
    for (int i = 0; i < NC; ++i) { v[i] = in[(size_t)r * NC + i]; mx = fmaxf(mx, v[i]); }
    float s = 0.f;
#pragma unroll
    for (int i = 0; i < NC; ++i) { v[i] = expf(v[i] - mx); s += v[i]; }
    float inv = 1.f / s;
#pragma unroll
    for (int i = 0; i < NC; ++i) outp[(size_t)r * NC + i] = v[i] * inv;
}

// ---------------------------------------------------------------------------
extern "C" void kernel_launch(void* const* d_in, const int* in_sizes, int n_in,
                              void* d_out, int out_size, void* d_ws, size_t ws_size,
                              hipStream_t stream)
{
    const int DE = 64, HID = 128;
    const int N = in_sizes[0] / DE;               // 50000
    const int E = in_sizes[1] / DE;               // 800000

    const float* x   = (const float*)d_in[0];
    const float* ef  = (const float*)d_in[1];
    const int*   ei  = (const int*)d_in[2];
    const int*   src = ei;
    const int*   dst = ei + E;
    const float* up_W   = (const float*)d_in[3];
    const float* up_b   = (const float*)d_in[4];
    const float* node_W = (const float*)d_in[5];
    const float* node_b = (const float*)d_in[6];
    const float* edge_W = (const float*)d_in[7];
    const float* edge_b = (const float*)d_in[8];
    const float* rec_W  = (const float*)d_in[9];
    const float* rec_b  = (const float*)d_in[10];

    // workspace carve-up (floats), 256B-aligned chunks
    float* base = (float*)d_ws;
    size_t off = 0;
    auto carve = [&](size_t n) { float* r = base + off; off += (n + 63) & ~(size_t)63; return r; };
    float*    h    = carve((size_t)N * HID);
    float*    ee   = carve((size_t)E * HID);
    float*    xl   = carve((size_t)N * HID);
    float*    xr   = carve((size_t)N * HID);
    float*    em   = carve((size_t)E * HID);
    float*    oacc = carve((size_t)N * HID);
    float*    evb  = carve((size_t)E * 8);
    unsigned* mxb  = (unsigned*)carve((size_t)N * 8);
    float*    den  = carve((size_t)N * 8);
    float*    nlog = carve((size_t)N * 8);
    float*    elog = carve((size_t)E * 6);
    (void)ws_size; (void)n_in; (void)out_size;

    float* out_node = (float*)d_out;              // [N,8]
    float* out_edge = out_node + (size_t)N * 8;   // [E,6]
    float* out_rec  = out_edge + (size_t)E * 6;   // [E,64]

    const dim3 blk(128);
    const dim3 gN64((N + 63) / 64, HID / 16);
    const dim3 gE64((E + 63) / 64, HID / 16);

    // up-projection (K=64): h = x@up_W+up_b ; ee = edge_features@up_W+up_b
    gemm_wmma<64><<<gN64, blk, 0, stream>>>(x,  nullptr, up_W, up_b, h,  N, HID);
    gemm_wmma<64><<<gE64, blk, 0, stream>>>(ef, nullptr, up_W, up_b, ee, E, HID);

    for (int b = 0; b < 2; ++b) {
        const float* Wl   = (const float*)d_in[11 + 9 * b];
        const float* bl   = (const float*)d_in[12 + 9 * b];
        const float* Wr   = (const float*)d_in[13 + 9 * b];
        const float* br   = (const float*)d_in[14 + 9 * b];
        const float* We   = (const float*)d_in[15 + 9 * b];
        const float* att  = (const float*)d_in[16 + 9 * b];
        const float* gbias= (const float*)d_in[17 + 9 * b];
        const float* ls   = (const float*)d_in[18 + 9 * b];
        const float* lb   = (const float*)d_in[19 + 9 * b];

        gemm_wmma<128><<<gN64, blk, 0, stream>>>(h,  nullptr, Wl, bl,      xl, N, HID);
        gemm_wmma<128><<<gN64, blk, 0, stream>>>(h,  nullptr, Wr, br,      xr, N, HID);
        gemm_wmma<128><<<gE64, blk, 0, stream>>>(ee, nullptr, We, nullptr, em, E, HID);

        hipMemsetAsync(mxb,  0, (size_t)N * 8   * sizeof(unsigned), stream);
        hipMemsetAsync(den,  0, (size_t)N * 8   * sizeof(float),    stream);
        hipMemsetAsync(oacc, 0, (size_t)N * HID * sizeof(float),    stream);

        int nEH = E * 8;
        edge_attn_logits<<<(nEH + 255) / 256, 256, 0, stream>>>(
            xl, xr, em, src, dst, att, evb, mxb, E);
        edge_exp_sum<<<(nEH + 255) / 256, 256, 0, stream>>>(evb, dst, mxb, den, E);
        int nEF = E * 128;
        edge_aggregate<<<(nEF + 255) / 256, 256, 0, stream>>>(
            evb, den, xl, src, dst, oacc, E);
        node_ln_relu<<<N, blk, 0, stream>>>(oacc, gbias, ls, lb, h);
    }

    // heads
    gemm_wmma<128><<<dim3((N + 63) / 64, 1), blk, 0, stream>>>(
        h, nullptr, node_W, node_b, nlog, N, 8);
    softmax_rows<8><<<(N + 255) / 256, 256, 0, stream>>>(nlog, out_node, N);

    gemm_wmma<128><<<dim3((E + 63) / 64, 1), blk, 0, stream>>>(
        h, src, edge_W, edge_b, elog, E, 6);
    softmax_rows<6><<<(E + 255) / 256, 256, 0, stream>>>(elog, out_edge, E);

    gemm_wmma<128><<<dim3((E + 63) / 64, 4), blk, 0, stream>>>(
        h, src, rec_W, rec_b, out_rec, E, 64);
}
